// EGCLayer_39006892982885
// MI455X (gfx1250) — matrix operand order repaired and checked by reading, not compiled, not run on hardware
//
#include <hip/hip_runtime.h>
#include <hip/hip_bf16.h>

typedef __bf16 v16bf __attribute__((ext_vector_type(16)));
typedef float  v8f   __attribute__((ext_vector_type(8)));

#define BB 8
#define NN 256
#define HH 128
#define JC 4      // j chunks
#define JCH 64    // j per chunk
#define WV 4      // waves per pair-kernel block

__device__ __forceinline__ float frcp(float x)     { return __builtin_amdgcn_rcpf(x); }
__device__ __forceinline__ float siluf(float x)    { return x * frcp(1.0f + __expf(-x)); }
__device__ __forceinline__ float sigmoidf_(float x){ return frcp(1.0f + __expf(-x)); }

// ---------------------------------------------------------------------------
// K0: pack edg2_w / cor2_w (128x128 f32 row-major) into WMMA B-fragment bf16
// layout: frag(ks,nt): lane l (group g=l>>4) element e holds W[ks*32+8g+e+(e>=8?8:0)][nt*16+(l&15)]
// ---------------------------------------------------------------------------
__global__ void k_pack_weights(const float* __restrict__ We, const float* __restrict__ Wc,
                               __bf16* __restrict__ Fe, __bf16* __restrict__ Fc) {
  int idx = blockIdx.x * blockDim.x + threadIdx.x;
  if (idx >= 2 * 16384) return;
  int m = idx >> 14;
  int t = idx & 16383;
  int e    = t & 15;
  int lane = (t >> 4) & 31;
  int nt   = (t >> 9) & 7;
  int ks   = t >> 12;
  int g = lane >> 4;
  int n = nt * 16 + (lane & 15);
  int k = ks * 32 + 8 * g + e + ((e >= 8) ? 8 : 0);
  const float* W = m ? Wc : We;
  __bf16* F = m ? Fc : Fe;
  F[t] = (__bf16)W[k * HH + n];
}

// ---------------------------------------------------------------------------
// K1: per-node layer-1 projections (layer-1 decomposition)
// ---------------------------------------------------------------------------
__global__ void k_proj(const float* __restrict__ h,
                       const float* __restrict__ e1w, const float* __restrict__ e1b,
                       const float* __restrict__ c1w, const float* __restrict__ c1b,
                       float* __restrict__ pea, float* __restrict__ peb,
                       float* __restrict__ pca, float* __restrict__ pcb) {
  int idx = blockIdx.x * blockDim.x + threadIdx.x;
  if (idx >= BB * NN * HH) return;
  int r = idx >> 7, c = idx & 127;
  const float* hr = h + r * HH;
  float a = 0.f, b = 0.f, p = 0.f, q = 0.f;
  #pragma unroll 4
  for (int k = 0; k < HH; ++k) {
    float hv = hr[k];
    a += hv * e1w[k * HH + c];
    b += hv * e1w[(HH + k) * HH + c];
    p += hv * c1w[k * HH + c];
    q += hv * c1w[(HH + k) * HH + c];
  }
  pea[idx] = a + e1b[c];
  peb[idx] = b;
  pca[idx] = p + c1b[c];
  pcb[idx] = q;
}

// ---------------------------------------------------------------------------
// K2: edge path over all pairs. Wave = one 16-i tile, loop over 64 j's.
// ---------------------------------------------------------------------------
__global__ __launch_bounds__(128) void k_edge_pairs(
    const float* __restrict__ x, const float* __restrict__ x0,
    const float* __restrict__ e1w,
    const float* __restrict__ pea, const float* __restrict__ peb,
    const __bf16* __restrict__ wfrag,
    const float* __restrict__ e2b,
    const float* __restrict__ eiw, const float* __restrict__ eib,
    float* __restrict__ pagg) {
  __shared__ float s_pea[WV][16][HH];
  __shared__ float s_xi[WV][16][3];
  __shared__ float s_x0i[WV][16][3];
  __shared__ float s_wd2[HH];
  __shared__ float s_wd02[HH];
  __shared__ float s_peb[HH];
  __shared__ float s_xj[3];
  __shared__ float s_x0j[3];

  const int tid  = threadIdx.x;
  const int b    = blockIdx.z;
  const int jc   = blockIdx.y;
  const int itg  = blockIdx.x;     // 0..3
  const int wave = tid >> 5;
  const int lane = tid & 31;
  const int mi   = lane & 15;
  const int g    = lane >> 4;

  for (int idx = tid; idx < WV * 16 * HH; idx += 128) {
    int w = idx >> 11;
    int rk = idx & 2047;
    int r = rk >> 7, k = rk & 127;
    int i = (itg * WV + w) * 16 + r;
    s_pea[w][r][k] = pea[(b * NN + i) * HH + k];
  }
  if (tid < HH) {
    s_wd2[tid]  = e1w[(2 * HH) * HH + tid];
    s_wd02[tid] = e1w[(2 * HH + 1) * HH + tid];
  }
  for (int idx = tid; idx < WV * 16 * 3; idx += 128) {
    int w = idx / 48; int rd = idx % 48; int r = rd / 3, d = rd % 3;
    int i = (itg * WV + w) * 16 + r;
    s_xi[w][r][d]  = x [(b * NN + i) * 3 + d];
    s_x0i[w][r][d] = x0[(b * NN + i) * 3 + d];
  }
  __syncthreads();

  const int itile = itg * WV + wave;
  const int ibase = itile * 16;
  float wi[8], be2[8];
  #pragma unroll
  for (int nt = 0; nt < 8; ++nt) { wi[nt] = eiw[nt * 16 + mi]; be2[nt] = e2b[nt * 16 + mi]; }
  const float bi = eib[0];

  // weight fragments resident in registers across the whole j loop
  const v16bf* wf = (const v16bf*)wfrag;
  v16bf bw[4][8];
  #pragma unroll
  for (int ks = 0; ks < 4; ++ks)
    #pragma unroll
    for (int nt = 0; nt < 8; ++nt)
      bw[ks][nt] = wf[(ks * 8 + nt) * 32 + lane];

  v8f agg[8];
  #pragma unroll
  for (int nt = 0; nt < 8; ++nt)
    #pragma unroll
    for (int v = 0; v < 8; ++v) agg[nt][v] = 0.f;

  for (int j = jc * JCH; j < jc * JCH + JCH; ++j) {
    __syncthreads();
    s_peb[tid] = peb[(b * NN + j) * HH + tid];          // blockDim==128==HH
    if (tid < 3)      s_xj[tid]      = x [(b * NN + j) * 3 + tid];
    else if (tid < 6) s_x0j[tid - 3] = x0[(b * NN + j) * 3 + (tid - 3)];
    __syncthreads();

    float d2 = 0.f, d02 = 0.f;
    #pragma unroll
    for (int d = 0; d < 3; ++d) {
      float dx = s_xi[wave][mi][d]  - s_xj[d];  d2  += dx * dx;
      float dy = s_x0i[wave][mi][d] - s_x0j[d]; d02 += dy * dy;
    }

    // A fragments: lane (g half) holds row mi, K per documented 16x32 layout
    v16bf A[4];
    #pragma unroll
    for (int ks = 0; ks < 4; ++ks) {
      #pragma unroll
      for (int e = 0; e < 16; ++e) {
        int k = ks * 32 + 8 * g + e + ((e >= 8) ? 8 : 0);
        float l1 = s_pea[wave][mi][k] + s_peb[k] + d2 * s_wd2[k] + d02 * s_wd02[k];
        A[ks][e] = (__bf16)siluf(l1);
      }
    }

    v8f m[8];
    #pragma unroll
    for (int nt = 0; nt < 8; ++nt) {
      v8f c;
      #pragma unroll
      for (int v = 0; v < 8; ++v) c[v] = 0.f;
      #pragma unroll
      for (int ks = 0; ks < 4; ++ks)
        c = __builtin_amdgcn_wmma_f32_16x16x32_bf16(false, A[ks], false, bw[ks][nt],
                                                    (short)0, c, false, false);
      m[nt] = c;
    }

    float s[8];
    #pragma unroll
    for (int v = 0; v < 8; ++v) s[v] = 0.f;
    #pragma unroll
    for (int nt = 0; nt < 8; ++nt)
      #pragma unroll
      for (int v = 0; v < 8; ++v) {
        float mv = siluf(m[nt][v] + be2[nt]);
        m[nt][v] = mv;
        s[v] += mv * wi[nt];
      }

    #pragma unroll
    for (int v = 0; v < 8; ++v) {
      float t = s[v];
      t += __shfl_xor(t, 1, 32);
      t += __shfl_xor(t, 2, 32);
      t += __shfl_xor(t, 4, 32);
      t += __shfl_xor(t, 8, 32);   // sum over 16-lane group (C rows v / v+8)
      int row = ibase + v + 8 * g;
      float ev = (row == j) ? 0.f : sigmoidf_(t + bi);
      #pragma unroll
      for (int nt = 0; nt < 8; ++nt) agg[nt][v] += ev * m[nt][v];
    }
  }

  #pragma unroll
  for (int nt = 0; nt < 8; ++nt)
    #pragma unroll
    for (int v = 0; v < 8; ++v) {
      int r = v + 8 * g;
      int col = nt * 16 + mi;
      pagg[(size_t)(jc * BB * NN + b * NN + ibase + r) * HH + col] = agg[nt][v];
    }
}

// ---------------------------------------------------------------------------
// K3: coordinate path over all pairs (same skeleton as K2).
// ---------------------------------------------------------------------------
__global__ __launch_bounds__(128) void k_cor_pairs(
    const float* __restrict__ x, const float* __restrict__ x0,
    const float* __restrict__ c1w,
    const float* __restrict__ pca, const float* __restrict__ pcb,
    const __bf16* __restrict__ wfrag,
    const float* __restrict__ c2b,
    const float* __restrict__ c3w, const float* __restrict__ c3b,
    float* __restrict__ pupd) {
  __shared__ float s_pca[WV][16][HH];
  __shared__ float s_xi[WV][16][3];
  __shared__ float s_x0i[WV][16][3];
  __shared__ float s_wd2[HH];
  __shared__ float s_wd02[HH];
  __shared__ float s_pcb[HH];
  __shared__ float s_xj[3];
  __shared__ float s_x0j[3];

  const int tid  = threadIdx.x;
  const int b    = blockIdx.z;
  const int jc   = blockIdx.y;
  const int itg  = blockIdx.x;
  const int wave = tid >> 5;
  const int lane = tid & 31;
  const int mi   = lane & 15;
  const int g    = lane >> 4;

  for (int idx = tid; idx < WV * 16 * HH; idx += 128) {
    int w = idx >> 11;
    int rk = idx & 2047;
    int r = rk >> 7, k = rk & 127;
    int i = (itg * WV + w) * 16 + r;
    s_pca[w][r][k] = pca[(b * NN + i) * HH + k];
  }
  if (tid < HH) {
    s_wd2[tid]  = c1w[(2 * HH) * HH + tid];
    s_wd02[tid] = c1w[(2 * HH + 1) * HH + tid];
  }
  for (int idx = tid; idx < WV * 16 * 3; idx += 128) {
    int w = idx / 48; int rd = idx % 48; int r = rd / 3, d = rd % 3;
    int i = (itg * WV + w) * 16 + r;
    s_xi[w][r][d]  = x [(b * NN + i) * 3 + d];
    s_x0i[w][r][d] = x0[(b * NN + i) * 3 + d];
  }
  __syncthreads();

  const int itile = itg * WV + wave;
  const int ibase = itile * 16;
  float wi[8], bc2[8];
  #pragma unroll
  for (int nt = 0; nt < 8; ++nt) { wi[nt] = c3w[nt * 16 + mi]; bc2[nt] = c2b[nt * 16 + mi]; }
  const float b3 = c3b[0];

  const v16bf* wf = (const v16bf*)wfrag;
  v16bf bw[4][8];
  #pragma unroll
  for (int ks = 0; ks < 4; ++ks)
    #pragma unroll
    for (int nt = 0; nt < 8; ++nt)
      bw[ks][nt] = wf[(ks * 8 + nt) * 32 + lane];

  float u0[8], u1[8], u2[8];
  #pragma unroll
  for (int v = 0; v < 8; ++v) { u0[v] = 0.f; u1[v] = 0.f; u2[v] = 0.f; }

  for (int j = jc * JCH; j < jc * JCH + JCH; ++j) {
    __syncthreads();
    s_pcb[tid] = pcb[(b * NN + j) * HH + tid];
    if (tid < 3)      s_xj[tid]      = x [(b * NN + j) * 3 + tid];
    else if (tid < 6) s_x0j[tid - 3] = x0[(b * NN + j) * 3 + (tid - 3)];
    __syncthreads();

    float d2 = 0.f, d02 = 0.f;
    #pragma unroll
    for (int d = 0; d < 3; ++d) {
      float dx = s_xi[wave][mi][d]  - s_xj[d];  d2  += dx * dx;
      float dy = s_x0i[wave][mi][d] - s_x0j[d]; d02 += dy * dy;
    }

    v16bf A[4];
    #pragma unroll
    for (int ks = 0; ks < 4; ++ks) {
      #pragma unroll
      for (int e = 0; e < 16; ++e) {
        int k = ks * 32 + 8 * g + e + ((e >= 8) ? 8 : 0);
        float l1 = s_pca[wave][mi][k] + s_pcb[k] + d2 * s_wd2[k] + d02 * s_wd02[k];
        A[ks][e] = (__bf16)siluf(l1);
      }
    }

    v8f m[8];
    #pragma unroll
    for (int nt = 0; nt < 8; ++nt) {
      v8f c;
      #pragma unroll
      for (int v = 0; v < 8; ++v) c[v] = 0.f;
      #pragma unroll
      for (int ks = 0; ks < 4; ++ks)
        c = __builtin_amdgcn_wmma_f32_16x16x32_bf16(false, A[ks], false, bw[ks][nt],
                                                    (short)0, c, false, false);
      m[nt] = c;
    }

    float s[8];
    #pragma unroll
    for (int v = 0; v < 8; ++v) s[v] = 0.f;
    #pragma unroll
    for (int nt = 0; nt < 8; ++nt)
      #pragma unroll
      for (int v = 0; v < 8; ++v)
        s[v] += siluf(m[nt][v] + bc2[nt]) * wi[nt];

    #pragma unroll
    for (int v = 0; v < 8; ++v) {
      float t = s[v];
      t += __shfl_xor(t, 1, 32);
      t += __shfl_xor(t, 2, 32);
      t += __shfl_xor(t, 4, 32);
      t += __shfl_xor(t, 8, 32);
      float cw = t + b3;
      int r = v + 8 * g;
      float dx0 = s_xi[wave][r][0] - s_xj[0];
      float dx1 = s_xi[wave][r][1] - s_xj[1];
      float dx2 = s_xi[wave][r][2] - s_xj[2];
      float d2r = dx0 * dx0 + dx1 * dx1 + dx2 * dx2;
      float dist = (d2r > 0.f) ? sqrtf(d2r) : 0.f;   // safe_dist: diag -> 0
      float inv = frcp(dist + 1.f);
      u0[v] += cw * dx0 * inv;   // diag: dx==0 -> contributes 0
      u1[v] += cw * dx1 * inv;
      u2[v] += cw * dx2 * inv;
    }
  }

  if (mi == 0) {                  // values uniform across 16-lane group
    #pragma unroll
    for (int v = 0; v < 8; ++v) {
      int r = v + 8 * g;
      size_t base = (size_t)(jc * BB * NN + b * NN + ibase + r) * 3;
      pupd[base + 0] = u0[v];
      pupd[base + 1] = u1[v];
      pupd[base + 2] = u2[v];
    }
  }
}

// ---------------------------------------------------------------------------
// K4a: deterministic fixed-order reduction of agg partials, /N
// ---------------------------------------------------------------------------
__global__ void k_agg(const float* __restrict__ pagg, float* __restrict__ agg) {
  int idx = blockIdx.x * blockDim.x + threadIdx.x;
  if (idx >= BB * NN * HH) return;
  const int SZ = BB * NN * HH;
  float s = pagg[idx] + pagg[SZ + idx] + pagg[2 * SZ + idx] + pagg[3 * SZ + idx];
  agg[idx] = s * (1.0f / NN);
}

// K4b: t = silu([h, agg] @ node1_w + b1)
__global__ void k_node1(const float* __restrict__ h, const float* __restrict__ agg,
                        const float* __restrict__ w1, const float* __restrict__ b1,
                        float* __restrict__ t) {
  int idx = blockIdx.x * blockDim.x + threadIdx.x;
  if (idx >= BB * NN * HH) return;
  int r = idx >> 7, c = idx & 127;
  float acc = b1[c];
  #pragma unroll 4
  for (int k = 0; k < HH; ++k) acc += h[r * HH + k] * w1[k * HH + c];
  #pragma unroll 4
  for (int k = 0; k < HH; ++k) acc += agg[r * HH + k] * w1[(HH + k) * HH + c];
  t[idx] = siluf(acc);
}

// K4c: h_next = (h + t @ node2_w + b2) * pad_mask
__global__ void k_node2(const float* __restrict__ h, const float* __restrict__ t,
                        const float* __restrict__ w2, const float* __restrict__ b2,
                        const float* __restrict__ pad, float* __restrict__ outh) {
  int idx = blockIdx.x * blockDim.x + threadIdx.x;
  if (idx >= BB * NN * HH) return;
  int r = idx >> 7, c = idx & 127;
  float acc = b2[c];
  #pragma unroll 4
  for (int k = 0; k < HH; ++k) acc += t[r * HH + k] * w2[k * HH + c];
  outh[idx] = (h[idx] + acc) * pad[r];
}

// K5: x_next = (x + mean_j upd) * pad_mask
__global__ void k_coord(const float* __restrict__ x, const float* __restrict__ pupd,
                        const float* __restrict__ pad, float* __restrict__ outx) {
  int idx = blockIdx.x * blockDim.x + threadIdx.x;
  if (idx >= BB * NN * 3) return;
  int r = idx / 3;
  const int SZ = BB * NN * 3;
  float s = pupd[idx] + pupd[SZ + idx] + pupd[2 * SZ + idx] + pupd[3 * SZ + idx];
  outx[idx] = (x[idx] + s * (1.0f / NN)) * pad[r];
}

// ---------------------------------------------------------------------------
extern "C" void kernel_launch(void* const* d_in, const int* in_sizes, int n_in,
                              void* d_out, int out_size, void* d_ws, size_t ws_size,
                              hipStream_t stream) {
  const float* x   = (const float*)d_in[0];
  const float* h   = (const float*)d_in[1];
  const float* x0  = (const float*)d_in[2];
  const float* pad = (const float*)d_in[3];
  const float* e1w = (const float*)d_in[4];
  const float* e1b = (const float*)d_in[5];
  const float* e2w = (const float*)d_in[6];
  const float* e2b = (const float*)d_in[7];
  const float* eiw = (const float*)d_in[8];
  const float* eib = (const float*)d_in[9];
  const float* n1w = (const float*)d_in[10];
  const float* n1b = (const float*)d_in[11];
  const float* n2w = (const float*)d_in[12];
  const float* n2b = (const float*)d_in[13];
  const float* c1w = (const float*)d_in[14];
  const float* c1b = (const float*)d_in[15];
  const float* c2w = (const float*)d_in[16];
  const float* c2b = (const float*)d_in[17];
  const float* c3w = (const float*)d_in[18];
  const float* c3b = (const float*)d_in[19];

  float* ws  = (float*)d_ws;
  float* pea = ws;                       // 262144 f32 each
  float* peb = pea + 262144;
  float* pca = peb + 262144;
  float* pcb = pca + 262144;
  float* agg = pcb + 262144;
  float* tn  = agg + 262144;
  __bf16* fe = (__bf16*)(tn + 262144);   // 16384 bf16 each, 32B-aligned
  __bf16* fc = fe + 16384;
  float* pagg = (float*)(fc + 16384);    // 4 * 262144 f32
  float* pupd = pagg + 4 * 262144;       // 4 * 6144 f32

  float* outx = (float*)d_out;           // (x_next, h_next) concat
  float* outh = outx + BB * NN * 3;

  k_pack_weights<<<dim3(128), dim3(256), 0, stream>>>(e2w, c2w, fe, fc);
  k_proj<<<dim3(1024), dim3(256), 0, stream>>>(h, e1w, e1b, c1w, c1b, pea, peb, pca, pcb);
  k_edge_pairs<<<dim3(4, JC, BB), dim3(128), 0, stream>>>(x, x0, e1w, pea, peb, fe, e2b, eiw, eib, pagg);
  k_cor_pairs<<<dim3(4, JC, BB), dim3(128), 0, stream>>>(x, x0, c1w, pca, pcb, fc, c2b, c3w, c3b, pupd);
  k_agg<<<dim3(1024), dim3(256), 0, stream>>>(pagg, agg);
  k_node1<<<dim3(1024), dim3(256), 0, stream>>>(h, agg, n1w, n1b, tn);
  k_node2<<<dim3(1024), dim3(256), 0, stream>>>(h, tn, n2w, n2b, pad, outh);
  k_coord<<<dim3(24), dim3(256), 0, stream>>>(x, pupd, pad, outx);
}
